// MyGATLayer_8452495638870
// MI455X (gfx1250) — matrix operand order repaired and checked by reading, not compiled, not run on hardware
//
#include <hip/hip_runtime.h>
#include <hip/hip_bf16.h>

// ---------------------------------------------------------------------------
// GAT forward for MI455X (gfx1250): bf16 WMMA projection + bandwidth-bound
// segment-softmax / scatter-add phases.
// ---------------------------------------------------------------------------

typedef __attribute__((ext_vector_type(16))) __bf16 v16bf;
typedef __attribute__((ext_vector_type(8)))  __bf16 v8bf;
typedef __attribute__((ext_vector_type(8)))  float  v8f;

#define IN_C   256
#define HD     256   // HEADS * OUT_C
#define HEADS  8

// ---- order-preserving float <-> int mapping for integer atomicMax ---------
__device__ __forceinline__ int f2ord(float f) {
    int i = __float_as_int(f);
    return (i >= 0) ? i : (i ^ 0x7fffffff);
}
__device__ __forceinline__ float ord2f(int i) {
    int j = (i >= 0) ? i : (i ^ 0x7fffffff);
    return __int_as_float(j);
}
// f2ord(-inf) == 0x807fffff
#define ORD_NEG_INF ((int)0x807fffff)

// ---------------------------------------------------------------------------
// Prep: fp32 -> bf16 copy of x
// ---------------------------------------------------------------------------
__global__ void k_convert_x(const float* __restrict__ x, __bf16* __restrict__ xb, int n) {
    int i = blockIdx.x * blockDim.x + threadIdx.x;
    int stride = gridDim.x * blockDim.x;
    for (; i < n; i += stride) xb[i] = (__bf16)x[i];
}

// Prep: W [K=IN_C][N=HD] fp32 -> WT [N][K] bf16 (so B fragments load contiguous)
__global__ void k_transpose_w(const float* __restrict__ W, __bf16* __restrict__ wt) {
    int k = blockIdx.x;    // 0..IN_C-1
    int c = threadIdx.x;   // 0..HD-1
    wt[c * IN_C + k] = (__bf16)W[k * HD + c];
}

// Init: zero output + softmax sums, -inf (ordered) for softmax max
__global__ void k_init(float* __restrict__ out, float* __restrict__ s,
                       int* __restrict__ m_ord, int n_nodes) {
    int i = blockIdx.x * blockDim.x + threadIdx.x;
    int stride = gridDim.x * blockDim.x;
    int total_out = n_nodes * HD;
    for (int j = i; j < total_out; j += stride) out[j] = 0.0f;
    int total_nh = n_nodes * HEADS;
    for (int j = i; j < total_nh; j += stride) { s[j] = 0.0f; m_ord[j] = ORD_NEG_INF; }
}

// ---------------------------------------------------------------------------
// GEMM: x_proj[N][256] = xb[N][256] @ W[256][256]  via v_wmma_f32_16x16x32_bf16
// Block = 128 threads (4 waves). Each wave: one 16(M) x 64(N) strip,
// 4 accumulators, A fragment reused across the 4 N-tiles, K unrolled 8x.
// ---------------------------------------------------------------------------
__global__ __launch_bounds__(128)
void k_gemm_wmma(const __bf16* __restrict__ xb, const __bf16* __restrict__ wt,
                 float* __restrict__ xproj, int n_rows) {
    const int lane = threadIdx.x & 31;
    const int wave = threadIdx.x >> 5;          // 0..3 -> N strip
    const int m0   = blockIdx.x * 16;
    const int c0   = wave * 64;
    const int l15  = lane & 15;
    const int hi   = lane >> 4;                 // K-half select per ISA layout

    int mrow = m0 + l15;
    if (mrow >= n_rows) mrow = n_rows - 1;      // clamp loads (stores guarded)

    v8f acc0 = {}, acc1 = {}, acc2 = {}, acc3 = {};

    const __bf16* __restrict__ arow = xb + (long)mrow * IN_C;

    #pragma unroll
    for (int k0 = 0; k0 < IN_C; k0 += 32) {
        // A fragment: 16-bit A 16x32 layout -> elems[0..7]=K(k0+hi*8..),
        // elems[8..15]=K(k0+16+hi*8..)
        v8bf alo = *(const v8bf*)(arow + k0 + hi * 8);
        v8bf ahi = *(const v8bf*)(arow + k0 + 16 + hi * 8);
        v16bf a = __builtin_shufflevector(alo, ahi,
                    0,1,2,3,4,5,6,7,8,9,10,11,12,13,14,15);

        const int kb = k0 + hi * 8;             // B: lane holds K(kb..kb+7), K(kb+16..kb+23) of col
        #pragma unroll
        for (int t = 0; t < 4; ++t) {
            const int col = c0 + t * 16 + l15;
            const __bf16* __restrict__ bp = wt + (long)col * IN_C + kb;
            v8bf b0 = *(const v8bf*)(bp);
            v8bf b1 = *(const v8bf*)(bp + 16);
            v16bf b = __builtin_shufflevector(b0, b1,
                        0,1,2,3,4,5,6,7,8,9,10,11,12,13,14,15);
            v8f cin = (t == 0) ? acc0 : (t == 1) ? acc1 : (t == 2) ? acc2 : acc3;
            v8f d = __builtin_amdgcn_wmma_f32_16x16x32_bf16(
                        false, a, false, b, (short)0, cin, false, false);
            if (t == 0) acc0 = d; else if (t == 1) acc1 = d;
            else if (t == 2) acc2 = d; else acc3 = d;
        }
    }

    // D layout: lane l, VGPR r -> M = r + 8*(l>=16), N = (l&15)
    const int rbase = m0 + hi * 8;
    #pragma unroll
    for (int r = 0; r < 8; ++r) {
        const int rr = rbase + r;
        if (rr < n_rows) {
            float* __restrict__ orow = xproj + (long)rr * HD + c0 + l15;
            orow[0]  = acc0[r];
            orow[16] = acc1[r];
            orow[32] = acc2[r];
            orow[48] = acc3[r];
        }
    }
}

// ---------------------------------------------------------------------------
// alpha_src[n][h] = dot(x_proj[n][h*32..], a_src[h]);  same for dst.
// One block per node; each wave32 covers exactly one head (32 dims).
// ---------------------------------------------------------------------------
__global__ __launch_bounds__(256)
void k_alpha(const float* __restrict__ xproj, const float* __restrict__ a_src,
             const float* __restrict__ a_dst, float* __restrict__ as,
             float* __restrict__ ad) {
    const int node = blockIdx.x;
    const int t = threadIdx.x;                  // 0..255 : h = t>>5, d = t&31
    const float v = xproj[(long)node * HD + t];
    float s1 = v * a_src[t];
    float s2 = v * a_dst[t];
    #pragma unroll
    for (int o = 16; o > 0; o >>= 1) {
        s1 += __shfl_xor(s1, o, 32);
        s2 += __shfl_xor(s2, o, 32);
    }
    if ((t & 31) == 0) {
        as[(long)node * HEADS + (t >> 5)] = s1;
        ad[(long)node * HEADS + (t >> 5)] = s2;
    }
}

// ---------------------------------------------------------------------------
// Segment max over incoming edges of each destination (ordered-int atomicMax)
// ---------------------------------------------------------------------------
__global__ void k_edge_max(const int* __restrict__ eidx, const float* __restrict__ as,
                           const float* __restrict__ ad, int* __restrict__ m_ord,
                           int ecnt) {
    const int* __restrict__ rowi = eidx;
    const int* __restrict__ coli = eidx + ecnt;
    int i = blockIdx.x * blockDim.x + threadIdx.x;
    int stride = gridDim.x * blockDim.x;
    for (int e = i; e < ecnt; e += stride) {
        const int r = rowi[e];
        const int c = coli[e];
        #pragma unroll
        for (int h = 0; h < HEADS; ++h) {
            float ev = as[r * HEADS + h] + ad[c * HEADS + h];
            atomicMax(&m_ord[c * HEADS + h], f2ord(ev));
        }
    }
}

// ---------------------------------------------------------------------------
// e_exp = exp(e - m[col]); s[col] += e_exp  (f32 atomics)
// ---------------------------------------------------------------------------
__global__ void k_edge_exp_sum(const int* __restrict__ eidx, const float* __restrict__ as,
                               const float* __restrict__ ad, const int* __restrict__ m_ord,
                               float* __restrict__ eexp, float* __restrict__ s, int ecnt) {
    const int* __restrict__ rowi = eidx;
    const int* __restrict__ coli = eidx + ecnt;
    int i = blockIdx.x * blockDim.x + threadIdx.x;
    int stride = gridDim.x * blockDim.x;
    for (int e = i; e < ecnt; e += stride) {
        const int r = rowi[e];
        const int c = coli[e];
        #pragma unroll
        for (int h = 0; h < HEADS; ++h) {
            float mv = ord2f(m_ord[c * HEADS + h]);   // finite: c has >=1 edge
            float ev = as[r * HEADS + h] + ad[c * HEADS + h];
            float ex = __expf(ev - mv);
            eexp[(long)e * HEADS + h] = ex;
            atomicAdd(&s[c * HEADS + h], ex);
        }
    }
}

// ---------------------------------------------------------------------------
// out[col][h*32+d] += (e_exp/(s[col]+1e-16)) * x_proj[row][h*32+d]
// One 256-thread block per edge; fully coalesced gather + f32 atomic scatter.
// ---------------------------------------------------------------------------
__global__ __launch_bounds__(256)
void k_edge_aggregate(const int* __restrict__ eidx, const float* __restrict__ eexp,
                      const float* __restrict__ s, const float* __restrict__ xproj,
                      float* __restrict__ out, int ecnt) {
    const int e = blockIdx.x;
    const int t = threadIdx.x;         // channel 0..255, head = t>>5
    const int h = t >> 5;
    const int r = eidx[e];
    const int c = eidx[ecnt + e];
    const float att = eexp[(long)e * HEADS + h] / (s[c * HEADS + h] + 1e-16f);
    atomicAdd(out + (long)c * HD + t, att * xproj[(long)r * HD + t]);
}

// ---------------------------------------------------------------------------
// Host launcher
// ---------------------------------------------------------------------------
extern "C" void kernel_launch(void* const* d_in, const int* in_sizes, int n_in,
                              void* d_out, int out_size, void* d_ws, size_t ws_size,
                              hipStream_t stream) {
    const float* x      = (const float*)d_in[0];
    const int*   eidx   = (const int*)d_in[1];
    const float* W      = (const float*)d_in[2];
    const float* a_src  = (const float*)d_in[3];
    const float* a_dst  = (const float*)d_in[4];
    float*       out    = (float*)d_out;

    const int n    = in_sizes[0] / IN_C;   // nodes
    const int ecnt = in_sizes[1] / 2;      // edges

    // Workspace carve-out (256B-aligned slices)
    char* ws = (char*)d_ws;
    size_t off = 0;
    auto take = [&](size_t bytes) -> void* {
        void* p = (void*)(ws + off);
        off += (bytes + 255) & ~(size_t)255;
        return p;
    };
    __bf16* xb    = (__bf16*)take((size_t)n * IN_C * 2);
    __bf16* wt    = (__bf16*)take((size_t)IN_C * HD * 2);
    float*  xproj = (float*) take((size_t)n * HD * 4);
    float*  as    = (float*) take((size_t)n * HEADS * 4);
    float*  ad    = (float*) take((size_t)n * HEADS * 4);
    int*    m_ord = (int*)   take((size_t)n * HEADS * 4);
    float*  ssum  = (float*) take((size_t)n * HEADS * 4);
    float*  eexp  = (float*) take((size_t)ecnt * HEADS * 4);
    (void)ws_size; (void)n_in; (void)out_size;

    // 1. fp32 -> bf16 conversions / transpose
    k_convert_x<<<2048, 256, 0, stream>>>(x, xb, n * IN_C);
    k_transpose_w<<<IN_C, HD, 0, stream>>>(W, wt);

    // 2. init out / softmax state
    k_init<<<2048, 256, 0, stream>>>(out, ssum, m_ord, n);

    // 3. projection GEMM via WMMA (16-row tiles; 50000 = 3125 * 16)
    k_gemm_wmma<<<(n + 15) / 16, 128, 0, stream>>>(xb, wt, xproj, n);

    // 4. attention logits per node/head
    k_alpha<<<n, 256, 0, stream>>>(xproj, a_src, a_dst, as, ad);

    // 5-7. segment softmax + weighted scatter aggregation
    k_edge_max<<<2048, 256, 0, stream>>>(eidx, as, ad, m_ord, ecnt);
    k_edge_exp_sum<<<2048, 256, 0, stream>>>(eidx, as, ad, m_ord, eexp, ssum, ecnt);
    k_edge_aggregate<<<ecnt, 256, 0, stream>>>(eidx, eexp, ssum, xproj, out, ecnt);
}
